// VectorQuantizer_82815559401688
// MI455X (gfx1250) — matrix operand order repaired and checked by reading, not compiled, not run on hardware
//
#include <hip/hip_runtime.h>
#include <math.h>

typedef __attribute__((ext_vector_type(16))) __bf16 v16bf;
typedef __attribute__((ext_vector_type(8)))  float  v8f;
typedef __attribute__((ext_vector_type(4)))  int    int4v;

#define NUM_CODES 2048
#define DIM 64
#define NROWS 65536
#define NTOT (NROWS*DIM)
#define NTILES (NUM_CODES/128)
#define FBIG 3.402823e38f

__device__ __forceinline__ void async_b128(const void* g, void* l) {
  __builtin_amdgcn_global_load_async_to_lds_b128(
      (__attribute__((address_space(1))) int4v*)g,
      (__attribute__((address_space(3))) int4v*)l, 0, 0);
}
__device__ __forceinline__ void async_b32(const void* g, void* l) {
  __builtin_amdgcn_global_load_async_to_lds_b32(
      (__attribute__((address_space(1))) int*)g,
      (__attribute__((address_space(3))) int*)l, 0, 0);
}

// ---------------------------------------------------------------------------
// K0: codebook prep. cnorm[k] = sum_d c[d][k]^2 ; split each element into
// bf16 hi/lo (c = hi + lo to ~fp32 accuracy) stored code-major for staging.
// Also zero-initializes histogram and loss accumulator (d_ws is poisoned).
// ---------------------------------------------------------------------------
__global__ void __launch_bounds__(256) vq_prep(
    const float* __restrict__ cb, float* __restrict__ cnorm,
    unsigned short* __restrict__ cbh, unsigned short* __restrict__ cbl,
    unsigned* __restrict__ hist, float* __restrict__ lossAccum)
{
  const int k = blockIdx.x * 256 + threadIdx.x;
  if (k >= NUM_CODES) return;
  float nrm = 0.f;
  for (int d = 0; d < DIM; ++d) {
    float c = cb[d * NUM_CODES + k];         // coalesced across k
    nrm += c * c;
    __bf16 hb = (__bf16)c;
    __bf16 lb = (__bf16)(c - (float)hb);
    cbh[k * DIM + d] = __builtin_bit_cast(unsigned short, hb);
    cbl[k * DIM + d] = __builtin_bit_cast(unsigned short, lb);
  }
  cnorm[k] = nrm;
  hist[k] = 0u;
  if (k == 0) lossAccum[0] = 0.f;
}

// ---------------------------------------------------------------------------
// K1: fused distance-GEMM (bf16x3 WMMA) + argmin + gather + losses + hist.
// Block = 256 threads (8 waves). Each wave owns 16 rows; A fragments stay in
// VGPRs across all 2048 codes. Codebook streamed in 128-code LDS tiles via
// double-buffered async global->LDS copies (ASYNCcnt). The argmin VALU is
// software-pipelined one subtile behind the WMMAs, so no WMMA->VALU hazard
// NOPs and the prev-score VALU hides LDS latency of the current B loads.
// ---------------------------------------------------------------------------
__global__ void __launch_bounds__(256) vq_main(
    const float* __restrict__ x, const float* __restrict__ cb,
    const float* __restrict__ cnorm,
    const unsigned short* __restrict__ cbh_g,
    const unsigned short* __restrict__ cbl_g,
    float* __restrict__ out, unsigned* __restrict__ hist,
    float* __restrict__ lossAccum)
{
  // padded row stride 72 halves (=144B, 16B aligned) to spread LDS banks
  __shared__ __align__(16) unsigned short lds_cbh[2][128 * 72];
  __shared__ __align__(16) unsigned short lds_cbl[2][128 * 72];
  __shared__ float    lds_cnorm[2][128];
  __shared__ unsigned lds_rowidx[8 * 16];

  const int tid   = threadIdx.x;
  const int lane  = tid & 31;
  const int wave  = tid >> 5;
  const int rowbase = blockIdx.x * 128 + wave * 16;
  const int n     = lane & 15;          // column / row-within-tile for this lane
  const int hi16  = (lane >> 4) & 1;    // upper half-wave?
  const int koffA = hi16 ? 16 : 0;      // A-matrix K offset (contiguous k order)
  const int koffB = hi16 ? 16 : 0;      // B-matrix K offset (contiguous k order)

  // ---- async stage of one 128-code tile (hi, lo, norms) into buffer `buf`
  auto stage_tile = [&](int tile, int buf) {
    const unsigned short* srcH = cbh_g + (size_t)tile * 128 * DIM;
    const unsigned short* srcL = cbl_g + (size_t)tile * 128 * DIM;
#pragma unroll
    for (int i = 0; i < 4; ++i) {
      int s     = (tid * 4 + i) * 4;    // dword index in tile (32 dwords/code)
      int code  = s >> 5;
      int dpair = s & 31;
      async_b128(srcH + s * 2, &lds_cbh[buf][code * 72 + dpair * 2]);
      async_b128(srcL + s * 2, &lds_cbl[buf][code * 72 + dpair * 2]);
    }
    if (tid < 128) async_b32(cnorm + tile * 128 + tid, &lds_cnorm[buf][tid]);
  };

  // ---- A fragments: row = rowbase + n, bf16 hi/lo, chunks K=[0,32),[32,64)
  const float* xrow = x + (size_t)(rowbase + n) * DIM;
  float av[32];
#pragma unroll
  for (int gi = 0; gi < 4; ++gi) {       // dims koffA + {0,8,32,40} + [0,8)
    float4 p0 = *reinterpret_cast<const float4*>(xrow + koffA + (gi >> 1) * 32 + (gi & 1) * 8);
    float4 p1 = *reinterpret_cast<const float4*>(xrow + koffA + (gi >> 1) * 32 + (gi & 1) * 8 + 4);
    av[gi*8+0]=p0.x; av[gi*8+1]=p0.y; av[gi*8+2]=p0.z; av[gi*8+3]=p0.w;
    av[gi*8+4]=p1.x; av[gi*8+5]=p1.y; av[gi*8+6]=p1.z; av[gi*8+7]=p1.w;
  }
  v16bf ah[2], al[2];
#pragma unroll
  for (int c = 0; c < 2; ++c)
#pragma unroll
    for (int h = 0; h < 16; ++h) {       // element h <-> K = c*32+koffA+h
      float v  = av[c * 16 + h];
      __bf16 hb = (__bf16)v;
      ah[c][h] = hb;
      al[c][h] = (__bf16)(v - (float)hb);
    }

  float    best[8];
  unsigned bidx[8];
#pragma unroll
  for (int r = 0; r < 8; ++r) { best[r] = FBIG; bidx[r] = 0u; }

  // pipelined "previous subtile" state; dummy init can never beat best
  v8f p0 = {}, p1 = {}, p2 = {};
  float    pcn   = FBIG;
  unsigned pcode = 0u;

  stage_tile(0, 0);                      // prologue

  for (int tile = 0; tile < NTILES; ++tile) {
    const int buf = tile & 1;
    __builtin_amdgcn_s_wait_asynccnt(0); // my share of tile `tile` landed
    __syncthreads();                     // everyone's landed; prev buf free
    if (tile + 1 < NTILES) stage_tile(tile + 1, buf ^ 1);

    // ---- 8 subtiles of 16 codes; 6 WMMAs each as 3 independent chains:
    //      a0 = xh.ch, a1 = xh.cl, a2 = xl.ch  (over both 32-dim K chunks).
    //      Score VALU is one subtile behind the WMMAs (software pipeline).
#pragma unroll 2
    for (int sub = 0; sub < 8; ++sub) {
      const int codeL = sub * 16 + n;
      const unsigned short* rowH = &lds_cbh[buf][codeL * 72];
      const unsigned short* rowL = &lds_cbl[buf][codeL * 72];
      // B element h <-> K = chunk*32 + koffB + h  (16 contiguous dims)
      const v16bf bh0 = *reinterpret_cast<const v16bf*>(rowH + koffB);
      const v16bf bl0 = *reinterpret_cast<const v16bf*>(rowL + koffB);
      const v16bf bh1 = *reinterpret_cast<const v16bf*>(rowH + 32 + koffB);
      const v16bf bl1 = *reinterpret_cast<const v16bf*>(rowL + 32 + koffB);
      v8f a0 = {}, a1 = {}, a2 = {};
      a0 = __builtin_amdgcn_wmma_f32_16x16x32_bf16(false, ah[0], false, bh0, (short)0, a0, false, false);
      a1 = __builtin_amdgcn_wmma_f32_16x16x32_bf16(false, ah[0], false, bl0, (short)0, a1, false, false);
      a2 = __builtin_amdgcn_wmma_f32_16x16x32_bf16(false, al[0], false, bh0, (short)0, a2, false, false);
      a0 = __builtin_amdgcn_wmma_f32_16x16x32_bf16(false, ah[1], false, bh1, (short)0, a0, false, false);
      a1 = __builtin_amdgcn_wmma_f32_16x16x32_bf16(false, ah[1], false, bl1, (short)0, a1, false, false);
      a2 = __builtin_amdgcn_wmma_f32_16x16x32_bf16(false, al[1], false, bh1, (short)0, a2, false, false);

      // process PREVIOUS subtile's scores (no dependence on the WMMAs above)
#pragma unroll
      for (int r = 0; r < 8; ++r) {      // dist = ||c||^2 - 2 x.c (row norm const)
        float dsc = fmaf(-2.0f, p0[r], pcn);
        dsc = fmaf(-2.0f, p1[r], dsc);
        dsc = fmaf(-2.0f, p2[r], dsc);
        if (dsc < best[r]) { best[r] = dsc; bidx[r] = pcode; }
      }

      p0 = a0; p1 = a1; p2 = a2;
      pcn   = lds_cnorm[buf][codeL];
      pcode = (unsigned)(tile * 128 + codeL);
    }
  }
  // ---- pipeline flush: last subtile's scores
#pragma unroll
  for (int r = 0; r < 8; ++r) {
    float dsc = fmaf(-2.0f, p0[r], pcn);
    dsc = fmaf(-2.0f, p1[r], dsc);
    dsc = fmaf(-2.0f, p2[r], dsc);
    if (dsc < best[r]) { best[r] = dsc; bidx[r] = pcode; }
  }

  // ---- per-row argmin across the 16 lanes sharing each row
#pragma unroll
  for (int r = 0; r < 8; ++r) {
    float    bv = best[r];
    unsigned bi = bidx[r];
#pragma unroll
    for (int m = 1; m <= 8; m <<= 1) {
      float    ov = __shfl_xor(bv, m, 32);
      unsigned oi = __shfl_xor(bi, m, 32);
      if (ov < bv || (ov == bv && oi < bi)) { bv = ov; bi = oi; }
    }
    if (n == r) {                        // lanes r and 16+r write rows r, r+8
      lds_rowidx[wave * 16 + r + hi16 * 8] = bi;
      atomicAdd(&hist[bi], 1u);
    }
  }
  __syncthreads();

  // ---- gather quantized vector, write ste (== quantized fwd), accumulate MSE
  const int      row  = rowbase + n;
  const unsigned qidx = lds_rowidx[wave * 16 + n];
  const int      db   = hi16 * 32;      // each lane handles 32 dims of its row
  float ls = 0.f;
#pragma unroll
  for (int v = 0; v < 8; ++v) {
    const int dim = db + v * 4;
    float4 xv = *reinterpret_cast<const float4*>(x + (size_t)row * DIM + dim);
    float4 q;
    q.x = cb[(dim + 0) * NUM_CODES + qidx];
    q.y = cb[(dim + 1) * NUM_CODES + qidx];
    q.z = cb[(dim + 2) * NUM_CODES + qidx];
    q.w = cb[(dim + 3) * NUM_CODES + qidx];
    *reinterpret_cast<float4*>(out + (size_t)row * DIM + dim) = q;
    float d0 = q.x - xv.x, d1 = q.y - xv.y, d2 = q.z - xv.z, d3 = q.w - xv.w;
    ls += d0*d0 + d1*d1 + d2*d2 + d3*d3;
  }
#pragma unroll
  for (int m = 1; m < 32; m <<= 1) ls += __shfl_xor(ls, m, 32);
  if (lane == 0) atomicAdd(lossAccum, ls);
}

// ---------------------------------------------------------------------------
// K2: perplexity from histogram + loss scalars.
// ---------------------------------------------------------------------------
__global__ void __launch_bounds__(256) vq_finalize(
    const unsigned* __restrict__ hist, const float* __restrict__ lossAccum,
    float* __restrict__ out)
{
  __shared__ float red[256];
  float s = 0.f;
  for (int k = threadIdx.x; k < NUM_CODES; k += 256) {
    float p = (float)hist[k] * (1.0f / (float)NROWS);
    s += p * logf(p + 1e-10f);
  }
  red[threadIdx.x] = s;
  __syncthreads();
  for (int off = 128; off > 0; off >>= 1) {
    if (threadIdx.x < off) red[threadIdx.x] += red[threadIdx.x + off];
    __syncthreads();
  }
  if (threadIdx.x == 0) {
    float mean = lossAccum[0] * (1.0f / (float)NTOT);
    out[NTOT + 0] = expf(-red[0]);   // perplexity
    out[NTOT + 1] = mean;            // codebook_loss
    out[NTOT + 2] = 0.25f * mean;    // commitment_loss
  }
}

extern "C" void kernel_launch(void* const* d_in, const int* in_sizes, int n_in,
                              void* d_out, int out_size, void* d_ws, size_t ws_size,
                              hipStream_t stream)
{
  const float* x  = (const float*)d_in[0];   // [64,32,32,64] f32
  const float* cb = (const float*)d_in[1];   // [64,2048] f32
  char* ws = (char*)d_ws;
  float*          cnorm = (float*)(ws);                            //   8 KB
  unsigned short* cbh   = (unsigned short*)(ws + 8192);            // 256 KB
  unsigned short* cbl   = (unsigned short*)(ws + 8192 + 262144);   // 256 KB
  unsigned*       hist  = (unsigned*)(ws + 8192 + 524288);         //   8 KB
  float*          lacc  = (float*)(ws + 8192 + 524288 + 8192);     //   4 B
  float* out = (float*)d_out;

  vq_prep    <<<NUM_CODES / 256, 256, 0, stream>>>(cb, cnorm, cbh, cbl, hist, lacc);
  vq_main    <<<NROWS / 128,     256, 0, stream>>>(x, cb, cnorm, cbh, cbl, out, hist, lacc);
  vq_finalize<<<1,               256, 0, stream>>>(hist, lacc, out);
}
